// iSQRTCOVPool_77489799954922
// MI455X (gfx1250) — matrix operand order repaired and checked by reading, not compiled, not run on hardware
//
#include <hip/hip_runtime.h>
#include <hip/hip_bf16.h>
#include <math.h>

typedef __attribute__((ext_vector_type(16))) _Float16 v16h;
typedef __attribute__((ext_vector_type(8)))  float    v8f;

#define Bc 32
#define Cc 256
#define Nn 3136           // 56*56 = 98 * 32
#define TRIU_PER_B 32896  // 256*257/2
#define LDA 40            // LDS row stride (halves): 80B rows -> 16B aligned frags
#define ATILE (64 * LDA)
#define BTILE (128 * LDA)

// ---------------------------------------------------------------------------
// Fragment loaders (contiguous k = 16*g + s per lane half, same for A and B)
// ---------------------------------------------------------------------------
__device__ __forceinline__ v16h load_fragA(const _Float16* rowbase, int g) {
  union { unsigned int u[8]; v16h v; } f;
  #pragma unroll
  for (int vv = 0; vv < 8; ++vv) {
    const int k = g * 16 + 2 * vv;
    f.u[vv] = *(const unsigned int*)(rowbase + k);
  }
  return f.v;
}
__device__ __forceinline__ v16h load_fragB(const _Float16* rowbase, int g) {
  union { unsigned int u[8]; v16h v; } f;
  #pragma unroll
  for (int vv = 0; vv < 8; ++vv)
    f.u[vv] = *(const unsigned int*)(rowbase + g * 16 + 2 * vv);
  return f.v;
}

// ---------------------------------------------------------------------------
// K1: per-(b,c) row sums: s = sum(x), sq = sum(x^2)
// ---------------------------------------------------------------------------
__global__ void rowsum_kernel(const float* __restrict__ x,
                              float* __restrict__ s, float* __restrict__ sq) {
  const int row = blockIdx.x;
  const float* xr = x + (size_t)row * Nn;
  float a0 = 0.f, a1 = 0.f;
  for (int i = threadIdx.x; i < Nn; i += blockDim.x) {
    float v = xr[i];
    a0 += v; a1 += v * v;
  }
  __shared__ float r0[256];
  __shared__ float r1[256];
  r0[threadIdx.x] = a0; r1[threadIdx.x] = a1;
  __syncthreads();
  for (int off = 128; off > 0; off >>= 1) {
    if ((int)threadIdx.x < off) {
      r0[threadIdx.x] += r0[threadIdx.x + off];
      r1[threadIdx.x] += r1[threadIdx.x + off];
    }
    __syncthreads();
  }
  if (threadIdx.x == 0) { s[row] = r0[0]; sq[row] = r1[0]; }
}

// ---------------------------------------------------------------------------
// K2: trace(cov)[b]
// ---------------------------------------------------------------------------
__global__ void trace_kernel(const float* __restrict__ s,
                             const float* __restrict__ sq,
                             float* __restrict__ tr) {
  const int b = blockIdx.x;
  const int c = threadIdx.x;
  const float n = (float)Nn;
  float sv = s[b * Cc + c];
  float qv = sq[b * Cc + c];
  float t = qv / n - (sv * sv) / (n * n);
  __shared__ float red[256];
  red[c] = t; __syncthreads();
  for (int off = 128; off > 0; off >>= 1) {
    if (c < off) red[c] += red[c + off];
    __syncthreads();
  }
  if (c == 0) tr[b] = red[0];
}

// ---------------------------------------------------------------------------
// K3: cov GEMM (double-buffered): 64x128 tile of cov = X X^T/n - s s^T/n^2;
//     fused epilogue: A = cov/tr, T = 0.5*(3I - A), both f16.
// ---------------------------------------------------------------------------
__global__ __launch_bounds__(128) void cov_kernel(
    const float* __restrict__ x, const float* __restrict__ s,
    const float* __restrict__ tr,
    _Float16* __restrict__ A_out, _Float16* __restrict__ T_out) {
  const int b  = blockIdx.z;
  const int i0 = blockIdx.y * 64;
  const int j0 = blockIdx.x * 128;
  const float* Xb = x + (size_t)b * Cc * Nn;

  __shared__ _Float16 As[2 * ATILE];
  __shared__ _Float16 Bs[2 * BTILE];

  const int t    = threadIdx.x;
  const int wave = t >> 5;
  const int lane = t & 31;
  const int g    = lane >> 4;
  const int lr   = lane & 15;
  const int wm   = (wave >> 1) * 32;   // wave tile: 32 x 64
  const int wn   = (wave & 1) * 64;

  v8f acc[8];
  #pragma unroll
  for (int i = 0; i < 8; ++i)
    #pragma unroll
    for (int j = 0; j < 8; ++j) acc[i][j] = 0.f;

  float4 av[4], bv[8];
  auto loadT = [&](int k0) {
    #pragma unroll
    for (int i = 0; i < 4; ++i) {
      const int q = t + i * 128, row = q >> 3, seg = q & 7;
      av[i] = *(const float4*)(Xb + (size_t)(i0 + row) * Nn + k0 + seg * 4);
    }
    #pragma unroll
    for (int i = 0; i < 8; ++i) {
      const int q = t + i * 128, row = q >> 3, seg = q & 7;
      bv[i] = *(const float4*)(Xb + (size_t)(j0 + row) * Nn + k0 + seg * 4);
    }
  };
  auto storeT = [&](_Float16* As_s, _Float16* Bs_s) {
    union { _Float16 h[4]; uint2 u; } p;
    #pragma unroll
    for (int i = 0; i < 4; ++i) {
      const int q = t + i * 128, row = q >> 3, seg = q & 7;
      p.h[0] = (_Float16)av[i].x; p.h[1] = (_Float16)av[i].y;
      p.h[2] = (_Float16)av[i].z; p.h[3] = (_Float16)av[i].w;
      *(uint2*)(&As_s[row * LDA + seg * 4]) = p.u;
    }
    #pragma unroll
    for (int i = 0; i < 8; ++i) {
      const int q = t + i * 128, row = q >> 3, seg = q & 7;
      p.h[0] = (_Float16)bv[i].x; p.h[1] = (_Float16)bv[i].y;
      p.h[2] = (_Float16)bv[i].z; p.h[3] = (_Float16)bv[i].w;
      *(uint2*)(&Bs_s[row * LDA + seg * 4]) = p.u;
    }
  };

  loadT(0);
  for (int step = 0; step < Nn / 32; ++step) {       // 98 steps
    _Float16* As_s = As + (step & 1) * ATILE;
    _Float16* Bs_s = Bs + (step & 1) * BTILE;
    storeT(As_s, Bs_s);
    __syncthreads();
    if (step < Nn / 32 - 1) loadT((step + 1) * 32);  // overlap with WMMA below

    v16h fa[2], fb[4];
    #pragma unroll
    for (int tm = 0; tm < 2; ++tm)
      fa[tm] = load_fragA(&As_s[(wm + tm * 16 + lr) * LDA], g);
    #pragma unroll
    for (int tn = 0; tn < 4; ++tn)
      fb[tn] = load_fragB(&Bs_s[(wn + tn * 16 + lr) * LDA], g);
    #pragma unroll
    for (int tm = 0; tm < 2; ++tm)
      #pragma unroll
      for (int tn = 0; tn < 4; ++tn)
        acc[tm * 4 + tn] = __builtin_amdgcn_wmma_f32_16x16x32_f16(
            false, fa[tm], false, fb[tn], (short)0, acc[tm * 4 + tn],
            false, false);
  }

  const float invn  = 1.0f / (float)Nn;
  const float invn2 = invn * invn;
  const float invtr = 1.0f / tr[b];
  const float* sb = s + b * Cc;
  #pragma unroll
  for (int tm = 0; tm < 2; ++tm)
    #pragma unroll
    for (int tn = 0; tn < 4; ++tn)
      #pragma unroll
      for (int r = 0; r < 8; ++r) {
        const int i = i0 + wm + tm * 16 + 8 * g + r;
        const int j = j0 + wn + tn * 16 + lr;
        const float cov = acc[tm * 4 + tn][r] * invn - sb[i] * sb[j] * invn2;
        const float a = cov * invtr;
        const size_t idx = ((size_t)b * Cc + i) * Cc + j;
        A_out[idx] = (_Float16)a;
        T_out[idx] = (_Float16)(0.5f * (((i == j) ? 3.0f : 0.0f) - a));
      }
}

// ---------------------------------------------------------------------------
// Shared body: 64x128 tile of 256^3 f16 GEMM, f32 accumulate, double-buffered.
// B staged TRANSPOSED (BT[n][k]) so B fragments are contiguous b128 reads.
// mode 0: D = P     mode 1: D = 0.5*(3I - P)
// ---------------------------------------------------------------------------
__device__ __forceinline__ void gemm_body_256(
    const _Float16* __restrict__ Ab, const _Float16* __restrict__ Bb,
    _Float16* __restrict__ Db, int i0, int j0, int mode,
    _Float16* As, _Float16* Bt) {
  const int t    = threadIdx.x;
  const int wave = t >> 5;
  const int lane = t & 31;
  const int g    = lane >> 4;
  const int lr   = lane & 15;
  const int wm   = (wave >> 1) * 32;
  const int wn   = (wave & 1) * 64;
  const int kk   = (t >> 4) * 4;      // this thread's 4 k-rows of the B tile
  const int nn   = (t & 15) * 8;      // and 8 n-cols

  v8f acc[8];
  #pragma unroll
  for (int i = 0; i < 8; ++i)
    #pragma unroll
    for (int j = 0; j < 8; ++j) acc[i][j] = 0.f;

  union U4 { uint4 v; unsigned short h[8]; };
  uint4 av[2];
  U4 bq[4];

  auto loadT = [&](int k0) {
    #pragma unroll
    for (int i = 0; i < 2; ++i) {
      const int q = t + i * 128, row = q >> 2, seg = q & 3;
      av[i] = *(const uint4*)(Ab + (size_t)(i0 + row) * Cc + k0 + seg * 8);
    }
    #pragma unroll
    for (int i = 0; i < 4; ++i)
      bq[i].v = *(const uint4*)(Bb + (size_t)(k0 + kk + i) * Cc + j0 + nn);
  };
  auto storeT = [&](_Float16* As_s, _Float16* Bt_s) {
    #pragma unroll
    for (int i = 0; i < 2; ++i) {
      const int q = t + i * 128, row = q >> 2, seg = q & 3;
      *(uint4*)(&As_s[row * LDA + seg * 8]) = av[i];
    }
    #pragma unroll
    for (int j = 0; j < 8; ++j) {   // transpose 4x8 sub-block in registers
      uint2 p;
      p.x = (unsigned int)bq[0].h[j] | ((unsigned int)bq[1].h[j] << 16);
      p.y = (unsigned int)bq[2].h[j] | ((unsigned int)bq[3].h[j] << 16);
      *(uint2*)(&Bt_s[(nn + j) * LDA + kk]) = p;
    }
  };

  loadT(0);
  for (int step = 0; step < Cc / 32; ++step) {       // 8 steps
    _Float16* As_s = As + (step & 1) * ATILE;
    _Float16* Bt_s = Bt + (step & 1) * BTILE;
    storeT(As_s, Bt_s);
    __syncthreads();
    if (step < Cc / 32 - 1) loadT((step + 1) * 32);

    v16h fa[2], fb[4];
    #pragma unroll
    for (int tm = 0; tm < 2; ++tm)
      fa[tm] = load_fragA(&As_s[(wm + tm * 16 + lr) * LDA], g);
    #pragma unroll
    for (int tn = 0; tn < 4; ++tn)
      fb[tn] = load_fragB(&Bt_s[(wn + tn * 16 + lr) * LDA], g);
    #pragma unroll
    for (int tm = 0; tm < 2; ++tm)
      #pragma unroll
      for (int tn = 0; tn < 4; ++tn)
        acc[tm * 4 + tn] = __builtin_amdgcn_wmma_f32_16x16x32_f16(
            false, fa[tm], false, fb[tn], (short)0, acc[tm * 4 + tn],
            false, false);
  }

  #pragma unroll
  for (int tm = 0; tm < 2; ++tm)
    #pragma unroll
    for (int tn = 0; tn < 4; ++tn)
      #pragma unroll
      for (int r = 0; r < 8; ++r) {
        const int i = i0 + wm + tm * 16 + 8 * g + r;
        const int j = j0 + wn + tn * 16 + lr;
        float p = acc[tm * 4 + tn][r];
        if (mode == 1) p = 0.5f * (((i == j) ? 3.0f : 0.0f) - p);
        Db[((size_t)i) * Cc + j] = (_Float16)p;
      }
}

// single GEMM (Y0 = A@T0 with mode 0; T = 0.5(3I - Z Y) with mode 1)
__global__ __launch_bounds__(128) void ns_gemm_kernel(
    const _Float16* __restrict__ Am, const _Float16* __restrict__ Bm,
    _Float16* __restrict__ Dm, int mode) {
  __shared__ _Float16 As[2 * ATILE];
  __shared__ _Float16 Bt[2 * BTILE];
  const int b = blockIdx.z;
  const size_t mo = (size_t)b * Cc * Cc;
  gemm_body_256(Am + mo, Bm + mo, Dm + mo, blockIdx.y * 64, blockIdx.x * 128,
                mode, As, Bt);
}

// fused pair: Yn = Y@T and Zn = T@Z in one launch (grid.z = 2*B)
__global__ __launch_bounds__(128) void ns_gemm_dual_kernel(
    const _Float16* __restrict__ Y, const _Float16* __restrict__ T,
    const _Float16* __restrict__ Z,
    _Float16* __restrict__ Yn, _Float16* __restrict__ Zn) {
  __shared__ _Float16 As[2 * ATILE];
  __shared__ _Float16 Bt[2 * BTILE];
  const int bz = blockIdx.z;
  const int b = bz >> 1;
  const size_t mo = (size_t)b * Cc * Cc;
  const _Float16* Am = (bz & 1) ? (T + mo) : (Y + mo);
  const _Float16* Bm = (bz & 1) ? (Z + mo) : (T + mo);
  _Float16*       Dm = (bz & 1) ? (Zn + mo) : (Yn + mo);
  gemm_body_256(Am, Bm, Dm, blockIdx.y * 64, blockIdx.x * 128, 0, As, Bt);
}

// ---------------------------------------------------------------------------
// K5: out[b, triu(r,c)] = Y[b,r,c] * sqrt(trace[b])
// ---------------------------------------------------------------------------
__global__ void triuvec_kernel(const _Float16* __restrict__ Y,
                               const float* __restrict__ tr,
                               float* __restrict__ out) {
  const int b = blockIdx.y;
  const int r = blockIdx.x;
  const int c = threadIdx.x;
  if (c < r) return;
  const float scale = sqrtf(tr[b]);
  const int rowStart = r * Cc - (r * (r - 1)) / 2;
  out[(size_t)b * TRIU_PER_B + rowStart + (c - r)] =
      (float)Y[((size_t)b * Cc + r) * Cc + c] * scale;
}

// ---------------------------------------------------------------------------
extern "C" void kernel_launch(void* const* d_in, const int* in_sizes, int n_in,
                              void* d_out, int out_size, void* d_ws,
                              size_t ws_size, hipStream_t stream) {
  const float* x = (const float*)d_in[0];
  float* out = (float*)d_out;

  char* ws = (char*)d_ws;
  size_t off = 0;
  auto carve = [&](size_t bytes) -> void* {
    void* p = ws + off;
    off = (off + bytes + 255) & ~(size_t)255;
    return p;
  };
  float* s  = (float*)carve((size_t)Bc * Cc * sizeof(float));
  float* sq = (float*)carve((size_t)Bc * Cc * sizeof(float));
  float* tr = (float*)carve((size_t)Bc * sizeof(float));
  _Float16* M[5];
  for (int i = 0; i < 5; ++i)
    M[i] = (_Float16*)carve((size_t)Bc * Cc * Cc * sizeof(_Float16));

  rowsum_kernel<<<Bc * Cc, 256, 0, stream>>>(x, s, sq);
  trace_kernel<<<Bc, 256, 0, stream>>>(s, sq, tr);

  const dim3 gridG(Cc / 128, Cc / 64, Bc);       // (2, 4, 32)
  const dim3 gridD(Cc / 128, Cc / 64, 2 * Bc);   // (2, 4, 64)

  // A = cov/tr (M0), T0 = 0.5*(3I - A) (M1)
  cov_kernel<<<gridG, 128, 0, stream>>>(x, s, tr, M[0], M[1]);
  // Y = A @ T0 ; Z = T0
  ns_gemm_kernel<<<gridG, 128, 0, stream>>>(M[0], M[1], M[2], 0);

  _Float16* Y = M[2];
  _Float16* Z = M[1];
  _Float16* freeb[3] = {M[0], M[3], M[4]};
  for (int it = 0; it < 4; ++it) {   // NUM_ITER - 1 = 4
    _Float16* T  = freeb[0];
    _Float16* Yn = freeb[1];
    _Float16* Zn = freeb[2];
    ns_gemm_kernel<<<gridG, 128, 0, stream>>>(Z, Y, T, 1);        // T = 0.5(3I-ZY)
    ns_gemm_dual_kernel<<<gridD, 128, 0, stream>>>(Y, T, Z, Yn, Zn);
    freeb[0] = Y; freeb[1] = Z; freeb[2] = T;
    Y = Yn; Z = Zn;
  }

  triuvec_kernel<<<dim3(Cc, Bc), 256, 0, stream>>>(Y, tr, out);
}
